// DynamicSparseAttention_53790170415658
// MI455X (gfx1250) — matrix operand order repaired and checked by reading, not compiled, not run on hardware
//
#include <hip/hip_runtime.h>
#include <hip/hip_bf16.h>

// DynamicSparseAttention on MI455X (gfx1250, wave32).
// B=2, S=1024, E=1024, H=16, hd=64, TOP_K=64.
//
// Pipeline (all on `stream`):
//   1) cvt_bf16:   x, Wq, Wk, Wv, Wo  fp32 -> bf16
//   2) gemm_wmma:  64x128 workgroup tile, 8 waves of 16x64; the shared
//      128-col x 32-k weight panel is staged in LDS per workgroup with
//      GLOBAL_LOAD_ASYNC_TO_LDS_B128 (ASYNCcnt double-buffered), WMMAs
//      run out of LDS for B and global b128 loads for A.
//      Q=(xWq^T+bq)/8 -> bf16 [B,H,S,64]; K -> bf16; V -> f32.
//   3) sparse_attn: WMMA score panel (16x1024) in LDS, exact top-64 by
//      16-bit bisection, softmax, weighted V gather.
//   4) gemm_wmma:  out = attn Wo^T + bo -> fp32 d_out
//
// Workspace (32 MiB):
//   [ 0) xb 4M | [4M) Wq 2M | [6M) Wk 2M | [8M) Wv 2M | [10M) Wo 2M
//   [12M) Qh bf16 4M | [16M) Kh bf16 4M | [20M) Vh f32 8M | [28M) attn bf16 4M

typedef __bf16        v16bf __attribute__((ext_vector_type(16)));
typedef float         v8f   __attribute__((ext_vector_type(8)));
typedef unsigned int  v4u   __attribute__((ext_vector_type(4)));

__device__ __forceinline__ unsigned short f2bf(float f) {
  unsigned u = __float_as_uint(f);
  unsigned r = 0x7fffu + ((u >> 16) & 1u);            // round to nearest even
  return (unsigned short)((u + r) >> 16);
}
__device__ __forceinline__ float bf2f(unsigned short h) {
  return __uint_as_float(((unsigned)h) << 16);
}

// A-matrix fragment (16x32 bf16): lane m=l&15, h=l>>4 holds
// K = h*16+{0..15}.  p points at (row m, K-offset kk).
__device__ __forceinline__ v16bf load_a_frag(const unsigned short* p, int hgrp) {
  union { v4u q[2]; v16bf v; } u;
  u.q[0] = *(const v4u*)(p + hgrp * 16);
  u.q[1] = *(const v4u*)(p + hgrp * 16 + 8);
  return u.v;
}

// ---------------------------------------------------------------------------
// fp32 -> bf16 convert
// ---------------------------------------------------------------------------
__global__ void cvt_bf16(const float* __restrict__ src,
                         unsigned short* __restrict__ dst, int n) {
  int i = blockIdx.x * blockDim.x + threadIdx.x;
  int stride = gridDim.x * blockDim.x;
  for (; i < n; i += stride) dst[i] = f2bf(src[i]);
}

// ---------------------------------------------------------------------------
// GEMM: out[m,n] = (sum_e A[m,e]*W[n,e] + bias[n]) * scale
// Fixed shape: M=2048, N=1024, K=1024.  Grid must be <<<256, 256>>>:
// blockIdx = tm (0..31, 64 rows) + 32*tn (0..7, 128 cols).
// 8 waves arranged 4(M) x 2(N); each wave owns a 16x64 tile (4 WMMA accs).
// Per 32-wide k-step the 128x32 B panel (8 KB) is async-DMA'd to LDS once
// per workgroup (2 GLOBAL_LOAD_ASYNC_TO_LDS_B128 per wave), double-buffered.
// mode 0: bf16 out, head-scatter [B,H,S,64]   (Q scale=1/8, K scale=1)
// mode 1: f32  out, head-scatter [B,H,S,64]   (V)
// mode 2: f32  out, plain [M x 1024]          (final projection)
// ---------------------------------------------------------------------------
__global__ __launch_bounds__(256) void gemm_wmma(
    const unsigned short* __restrict__ A,
    const unsigned short* __restrict__ W,
    const float* __restrict__ bias,
    void* __restrict__ outp, int mode, float scale) {
  __shared__ __attribute__((aligned(64))) unsigned short bpanel[2][128 * 32];

  const int tid  = threadIdx.x;
  const int lane = tid & 31;
  const int w    = tid >> 5;           // wave 0..7
  const int wm   = w & 3;              // M sub-tile within workgroup
  const int wn   = w >> 2;             // N sub-tile (0/1)
  const int tm   = blockIdx.x & 31;    // 32 tiles of 64 rows
  const int tn   = blockIdx.x >> 5;    // 8 tiles of 128 cols
  const int mrow = lane & 15;
  const int hgrp = lane >> 4;
  const int n0   = tn * 128;

  const unsigned short* arow =
      A + (size_t)(tm * 64 + wm * 16 + mrow) * 1024;

  // Stage the 128x32 bf16 panel for k-offset kk into bpanel[buf].
  // 512 chunks of 8 elems (16 B); thread t moves chunks 2t, 2t+1.
  auto stage = [&](int buf, int kk) {
#pragma unroll
    for (int e = 0; e < 2; ++e) {
      const int q  = 2 * tid + e;
      const int c  = q >> 2;           // panel column 0..127
      const int j8 = (q & 3) * 8;      // k-offset within panel
      const unsigned short* g = W + (size_t)(n0 + c) * 1024 + kk + j8;
      const unsigned lds =
          (unsigned)(uintptr_t)&bpanel[buf][c * 32 + j8];
      asm volatile("global_load_async_to_lds_b128 %0, %1, off"
                   :: "v"(lds), "v"((unsigned long long)(uintptr_t)g)
                   : "memory");
    }
  };

  v8f acc[4];
#pragma unroll
  for (int t = 0; t < 4; ++t) acc[t] = (v8f){0.f,0.f,0.f,0.f,0.f,0.f,0.f,0.f};

  stage(0, 0);
  for (int kk = 0; kk < 1024; kk += 32) {
    const int buf = (kk >> 5) & 1;
    if (kk + 32 < 1024) {
      stage(buf ^ 1, kk + 32);
      // retire this buffer's 2 async ops; next buffer's 2 stay in flight
      asm volatile("s_wait_asynccnt 2" ::: "memory");
    } else {
      asm volatile("s_wait_asynccnt 0" ::: "memory");
    }
    __syncthreads();                   // panel visible workgroup-wide

    __builtin_prefetch(arow + kk + 128, 0, 1);   // global_prefetch_b8
    v16bf a = load_a_frag(arow + kk, hgrp);
#pragma unroll
    for (int t = 0; t < 4; ++t) {
      const unsigned short* bp =
          &bpanel[buf][(wn * 64 + t * 16 + mrow) * 32 + hgrp * 16];
      v16bf b = *(const v16bf*)bp;     // 32 B from LDS (ds_load)
      acc[t] = __builtin_amdgcn_wmma_f32_16x16x32_bf16(
          false, a, false, b, (short)0, acc[t], false, false);
    }
    __syncthreads();                   // protect buffer before next overwrite
  }

  unsigned short* obf = (unsigned short*)outp;
  float*          of  = (float*)outp;
#pragma unroll
  for (int t = 0; t < 4; ++t) {
    const int ng = n0 + wn * 64 + t * 16 + mrow;      // global column
    const float bv = bias[ng];
#pragma unroll
    for (int r = 0; r < 8; ++r) {                     // C layout: M = r+8*hgrp
      const int mg = tm * 64 + wm * 16 + r + 8 * hgrp;
      const float y = (acc[t][r] + bv) * scale;
      if (mode == 2) {
        of[(size_t)mg * 1024 + ng] = y;
      } else {
        const int bb = mg >> 10, s = mg & 1023;
        const int hh = ng >> 6,  d = ng & 63;
        const size_t off = ((size_t)((bb * 16 + hh) * 1024 + s)) * 64 + d;
        if (mode == 0) obf[off] = f2bf(y);
        else           of[off]  = y;
      }
    }
  }
}

// ---------------------------------------------------------------------------
// Sparse attention: one wave (32 threads) per 16-query tile of one (b,h).
// Scores via WMMA into a bf16 LDS panel; exact 64th-largest threshold via
// bisection on order-preserving u16 keys; softmax + weighted V gather.
// ---------------------------------------------------------------------------
__global__ __launch_bounds__(32) void sparse_attn(
    const unsigned short* __restrict__ qh,   // [32][1024][64] bf16 (pre /8)
    const unsigned short* __restrict__ kh,   // [32][1024][64] bf16
    const float* __restrict__ vh,            // [32][1024][64] f32
    unsigned short* __restrict__ attnb) {    // [2][1024][1024] bf16
  __shared__ unsigned short sc[16 * 1024];   // 32 KiB score panel
  __shared__ unsigned int   sidx[64];
  __shared__ float          swt[64];

  const int lane = threadIdx.x;
  const int tile = blockIdx.x;               // 0..2047
  const int bh   = tile >> 6;                // 0..31 = b*16 + h
  const int q0   = (tile & 63) << 4;
  const int mrow = lane & 15;
  const int hgrp = lane >> 4;

  const unsigned short* qbase = qh + ((size_t)bh * 1024 + q0) * 64;
  const unsigned short* kbase = kh + (size_t)bh * 1024 * 64;

  const v16bf a0 = load_a_frag(qbase + (size_t)mrow * 64 + 0,  hgrp);
  const v16bf a1 = load_a_frag(qbase + (size_t)mrow * 64 + 32, hgrp);

  for (int kt = 0; kt < 64; ++kt) {
    const unsigned short* kb =
        kbase + (size_t)(kt * 16 + mrow) * 64 + hgrp * 16;
    v16bf b0 = *(const v16bf*)(kb);          // head-dim 0..31 slice
    v16bf b1 = *(const v16bf*)(kb + 32);     // head-dim 32..63 slice
    v8f c = (v8f){0.f,0.f,0.f,0.f,0.f,0.f,0.f,0.f};
    c = __builtin_amdgcn_wmma_f32_16x16x32_bf16(false, a0, false, b0,
                                                (short)0, c, false, false);
    c = __builtin_amdgcn_wmma_f32_16x16x32_bf16(false, a1, false, b1,
                                                (short)0, c, false, false);
#pragma unroll
    for (int r = 0; r < 8; ++r)
      sc[(r + 8 * hgrp) * 1024 + kt * 16 + mrow] = f2bf(c[r]);
  }
  __syncthreads();

  const int b = bh >> 4, h = bh & 15;

  for (int q = 0; q < 16; ++q) {
    const unsigned short* row = sc + q * 1024;

    unsigned short keyv[32];
    float lmax = -3.0e38f;
#pragma unroll
    for (int i = 0; i < 32; ++i) {
      const unsigned short raw = row[i * 32 + lane];
      keyv[i] = (raw & 0x8000u) ? (unsigned short)(~raw)
                                : (unsigned short)(raw | 0x8000u);
      lmax = fmaxf(lmax, bf2f(raw));
    }
#pragma unroll
    for (int off = 16; off > 0; off >>= 1)
      lmax = fmaxf(lmax, __shfl_xor(lmax, off, 32));

    // bisection: largest thr with count(key >= thr) >= 64 (== 64th largest)
    unsigned thr = 0;
    for (int bit = 15; bit >= 0; --bit) {
      const unsigned cand = thr | (1u << bit);
      int cnt = 0;
#pragma unroll
      for (int i = 0; i < 32; ++i) cnt += (keyv[i] >= cand) ? 1 : 0;
#pragma unroll
      for (int off = 16; off > 0; off >>= 1) cnt += __shfl_xor(cnt, off, 32);
      if (cnt >= 64) thr = cand;
    }

    // ballot-prefix compaction of selected (idx, exp(s-max)) pairs
    int total = 0;
#pragma unroll
    for (int i = 0; i < 32; ++i) {
      const bool sel = keyv[i] >= thr;
      const unsigned mask = __builtin_amdgcn_ballot_w32(sel);
      const int rank = total + __popc(mask & ((1u << lane) - 1u));
      if (sel && rank < 64) {
        const int kidx = i * 32 + lane;
        sidx[rank] = (unsigned)kidx;
        swt[rank]  = __expf(bf2f(row[kidx]) - lmax);
      }
      total += __popc(mask);
    }
    const int nsel = total < 64 ? total : 64;
    __syncthreads();

    float dsum = 0.f;
    for (int j = lane; j < nsel; j += 32) dsum += swt[j];
#pragma unroll
    for (int off = 16; off > 0; off >>= 1) dsum += __shfl_xor(dsum, off, 32);
    const float inv = 1.0f / dsum;

    float o0 = 0.f, o1 = 0.f;
    for (int j = 0; j < nsel; ++j) {
      const float wj = swt[j];
      const float* vrow = vh + ((size_t)bh * 1024 + sidx[j]) * 64;
      o0 += wj * vrow[lane];
      o1 += wj * vrow[lane + 32];
    }
    o0 *= inv; o1 *= inv;

    const int s = q0 + q;
    const size_t obase = ((size_t)(b * 1024 + s)) * 1024 + h * 64;
    attnb[obase + lane]      = f2bf(o0);
    attnb[obase + lane + 32] = f2bf(o1);
    __syncthreads();
  }
}

// ---------------------------------------------------------------------------
extern "C" void kernel_launch(void* const* d_in, const int* in_sizes, int n_in,
                              void* d_out, int out_size, void* d_ws,
                              size_t ws_size, hipStream_t stream) {
  (void)in_sizes; (void)n_in; (void)out_size; (void)ws_size;
  const float* x  = (const float*)d_in[0];
  const float* Wq = (const float*)d_in[1];
  const float* bq = (const float*)d_in[2];
  const float* Wk = (const float*)d_in[3];
  const float* bk = (const float*)d_in[4];
  const float* Wv = (const float*)d_in[5];
  const float* bv = (const float*)d_in[6];
  const float* Wo = (const float*)d_in[7];
  const float* bo = (const float*)d_in[8];

  unsigned char* ws = (unsigned char*)d_ws;
  unsigned short* xb  = (unsigned short*)(ws + (size_t)0);
  unsigned short* wqb = (unsigned short*)(ws + ((size_t)4  << 20));
  unsigned short* wkb = (unsigned short*)(ws + ((size_t)6  << 20));
  unsigned short* wvb = (unsigned short*)(ws + ((size_t)8  << 20));
  unsigned short* wob = (unsigned short*)(ws + ((size_t)10 << 20));
  unsigned short* qhb = (unsigned short*)(ws + ((size_t)12 << 20));
  unsigned short* khb = (unsigned short*)(ws + ((size_t)16 << 20));
  float*          vhf = (float*)         (ws + ((size_t)20 << 20));
  unsigned short* atb = (unsigned short*)(ws + ((size_t)28 << 20));

  const int XE = 2048 * 1024;
  const int WE = 1024 * 1024;

  cvt_bf16<<<512, 256, 0, stream>>>(x,  xb,  XE);
  cvt_bf16<<<512, 256, 0, stream>>>(Wq, wqb, WE);
  cvt_bf16<<<512, 256, 0, stream>>>(Wk, wkb, WE);
  cvt_bf16<<<512, 256, 0, stream>>>(Wv, wvb, WE);
  cvt_bf16<<<512, 256, 0, stream>>>(Wo, wob, WE);

  // 32 M-tiles x 8 N-tiles = 256 workgroups of 8 waves
  gemm_wmma<<<256, 256, 0, stream>>>(xb, wqb, bq, qhb, 0, 0.125f);
  gemm_wmma<<<256, 256, 0, stream>>>(xb, wkb, bk, khb, 0, 1.0f);
  gemm_wmma<<<256, 256, 0, stream>>>(xb, wvb, bv, vhf, 1, 1.0f);

  sparse_attn<<<2048, 32, 0, stream>>>(qhb, khb, vhf, atb);

  gemm_wmma<<<256, 256, 0, stream>>>(atb, wob, bo, d_out, 2, 1.0f);
}
